// encoderBlock_weight_demod_36275293782588
// MI455X (gfx1250) — matrix operand order repaired and checked by reading, not compiled, not run on hardware
//
#include <hip/hip_runtime.h>
#include <hip/hip_bf16.h>

typedef __bf16 bf16_t;
typedef __attribute__((ext_vector_type(16))) __bf16 v16bf;
typedef __attribute__((ext_vector_type(8)))  float  v8f;
typedef __attribute__((ext_vector_type(4)))  unsigned int u32x4;

#define LRELU_SLOPE 0.2f

// A-matrix fragment (16x32 bf16): lane (M row) holds 16 contiguous K values
// starting at K = half*16. p is pre-offset by (kstep*32 + half*16).
__device__ __forceinline__ v16bf ld_a(const bf16_t* p) {
  union { u32x4 q[2]; v16bf v; } u;
  u.q[0] = *(const u32x4*)(p);
  u.q[1] = *(const u32x4*)(p + 8);
  return u.v;
}

// B-matrix fragment (32x16 bf16): lane (N col) holds 16 contiguous K values
// starting at K = half*16. p is pre-offset by (kstep*32 + half*16).
__device__ __forceinline__ v16bf ld_b(const bf16_t* p) {
  union { u32x4 q[2]; v16bf v; } u;
  u.q[0] = *(const u32x4*)(p);
  u.q[1] = *(const u32x4*)(p + 8);
  return u.v;
}

__device__ __forceinline__ v8f wmma_bf16(v16bf a, v16bf b, v8f c) {
  return __builtin_amdgcn_wmma_f32_16x16x32_bf16(false, a, false, b, (short)0, c,
                                                 false, false);
}

// ---------------------------------------------------------------------------
// StyleGAN2 weight modulate + demodulate: one block per (b, c_out).
// w: [128][128][3][3] f32, style: [B][128] f32 -> wmod bf16 [b][tap9][cout][cin]
// ---------------------------------------------------------------------------
__global__ __launch_bounds__(256) void modulate_kernel(
    const float* __restrict__ w, const float* __restrict__ style,
    bf16_t* __restrict__ wmod) {
  const int cout = blockIdx.x & 127;
  const int b    = blockIdx.x >> 7;
  __shared__ float wsh[1152];
  __shared__ float red[256];
  float ss = 0.0f;
  for (int i = threadIdx.x; i < 1152; i += 256) {
    const int cin = i / 9;
    const float v = w[cout * 1152 + i] * (style[b * 128 + cin] + 1.0f);
    wsh[i] = v;
    ss += v * v;
  }
  red[threadIdx.x] = ss;
  __syncthreads();
  for (int s = 128; s > 0; s >>= 1) {
    if (threadIdx.x < s) red[threadIdx.x] += red[threadIdx.x + s];
    __syncthreads();
  }
  const float d = rsqrtf(red[0] + 1e-8f);
  for (int i = threadIdx.x; i < 1152; i += 256) {
    const int cin = i / 9;
    const int t   = i % 9;
    wmod[(((size_t)b * 9 + t) * 128 + cout) * 128 + cin] = (bf16_t)(wsh[i] * d);
  }
}

// down_w [256][128][4][4] f32 -> bf16 [tap16][cout256][cin128]
__global__ __launch_bounds__(256) void cvt_downw_kernel(
    const float* __restrict__ w, bf16_t* __restrict__ o) {
  const int i = blockIdx.x * 256 + threadIdx.x;
  if (i < 256 * 128 * 16) {
    const int t    = i & 15;
    const int cin  = (i >> 4) & 127;
    const int cout = i >> 11;
    o[((size_t)t * 256 + cout) * 128 + cin] = (bf16_t)w[i];
  }
}

// ---------------------------------------------------------------------------
// 3x3 SAME conv as implicit GEMM, per-batch modulated bf16 weights.
// Workgroup: (b, 4-row y-block, 32-pixel x-tile). 8 waves x (M=16, N=32, 4 rows).
// LDS: 6 rows x 34 cols x 128 ch (52 KB), channel-innermost so each B frag is
// two ds_load_b128. Each A frag feeds 8 WMMAs (4 rows x 2 n-tiles).
// 9 taps x 4 ksteps x 8 = 288 WMMAs / wave. Residual read from LDS tile.
// ---------------------------------------------------------------------------
template <bool IN_F32, bool RESID>
__global__ __launch_bounds__(256) void conv3x3_mod_kernel(
    const float* __restrict__ inF, const bf16_t* __restrict__ inB,
    const bf16_t* __restrict__ wmod, bf16_t* __restrict__ outB) {
  constexpr int H = 128, W = 128, C = 128, ROWS = 4;
  const int bid = blockIdx.x;
  const int xt  = bid & 3;           // W/32
  const int yt  = (bid >> 2) & 31;   // H/ROWS
  const int b   = bid >> 7;
  const int x0  = xt * 32;
  const int y0  = yt * ROWS;

  __shared__ bf16_t smem[(ROWS + 2) * 34 * 128];  // 52224 B

  const int tid = threadIdx.x;
  for (int i = tid; i < (ROWS + 2) * 34 * 128; i += 256) {
    const int col = i % 34;
    const int ch  = (i / 34) & 127;
    const int row = i / (34 * 128);
    const int gx = x0 + col - 1;
    const int gy = y0 + row - 1;
    float v = 0.0f;
    if (gx >= 0 && gx < W && gy >= 0 && gy < H) {
      const size_t g = (((size_t)b * C + ch) * H + gy) * W + gx;
      v = IN_F32 ? inF[g] : (float)inB[g];
    }
    smem[(row * 34 + col) * 128 + ch] = (bf16_t)v;
  }
  __syncthreads();

  const int wave = tid >> 5;
  const int lane = tid & 31;
  const int half = lane >> 4;
  const int mn   = lane & 15;   // M row (A) / N col (B)
  const int m0   = wave * 16;

  v8f acc[ROWS][2];
#pragma unroll
  for (int r = 0; r < ROWS; ++r) {
    acc[r][0] = (v8f){};
    acc[r][1] = (v8f){};
  }

  const bf16_t* wb = wmod + (size_t)b * 9 * 128 * 128;
  for (int t = 0; t < 9; ++t) {
    const int dy = t / 3, dx = t % 3;
    const bf16_t* wr = wb + ((size_t)t * 128 + (m0 + mn)) * 128;
#pragma unroll
    for (int ks = 0; ks < 4; ++ks) {
      const int ka = ks * 32 + half * 16;
      const int kb = ks * 32 + half * 16;
      const v16bf a = ld_a(wr + ka);
#pragma unroll
      for (int r = 0; r < ROWS; ++r) {
        // output row y0+r needs input row (y0+r-1+dy) -> LDS row (r+dy)
        const bf16_t* bp0 = &smem[((r + dy) * 34 + (mn + dx)) * 128];
        const bf16_t* bp1 = &smem[((r + dy) * 34 + (mn + 16 + dx)) * 128];
        acc[r][0] = wmma_bf16(a, ld_b(bp0 + kb), acc[r][0]);
        acc[r][1] = wmma_bf16(a, ld_b(bp1 + kb), acc[r][1]);
      }
    }
  }

#pragma unroll
  for (int r = 0; r < ROWS; ++r) {
    const int y = y0 + r;
#pragma unroll
    for (int e = 0; e < 8; ++e) {
      const int co = m0 + e + half * 8;  // C/D layout: M = e + half*8
      const size_t base = (((size_t)b * C + co) * H + y) * W;
      const int px0 = x0 + mn;
      const int px1 = x0 + 16 + mn;
      float v0 = acc[r][0][e];
      float v1 = acc[r][1][e];
      if (RESID) {
        // residual x[b][co][y][px] lives in the staged tile (C_out == C_in)
        v0 += (float)smem[((r + 1) * 34 + (mn + 1)) * 128 + co];
        v1 += (float)smem[((r + 1) * 34 + (mn + 17)) * 128 + co];
      }
      v0 = v0 >= 0.0f ? v0 : LRELU_SLOPE * v0;
      v1 = v1 >= 0.0f ? v1 : LRELU_SLOPE * v1;
      outB[base + px0] = (bf16_t)v0;
      outB[base + px1] = (bf16_t)v1;
    }
  }
}

// ---------------------------------------------------------------------------
// 4x4 stride-2 pad-1 conv, 128 -> 256 ch, + bias, f32 out.
// Workgroup: (b, 2-row oy-block, 16-pixel ox-tile, 128-ch m-group).
// LDS: 6 rows x 36 cols x 128 ch (54 KB). 16 taps x 4 ksteps x 2 rows
// = 128 WMMAs / wave; each A frag feeds 2 WMMAs.
// ---------------------------------------------------------------------------
__global__ __launch_bounds__(256) void conv4x4s2_kernel(
    const bf16_t* __restrict__ xin, const bf16_t* __restrict__ wd,
    const float* __restrict__ bias, float* __restrict__ out) {
  constexpr int H = 128, W = 128, C = 128, CO = 256, HO = 64, WO = 64, ROWS = 2;
  const int bid = blockIdx.x;
  const int mg  = bid & 1;
  const int xt  = (bid >> 1) & 3;
  const int oyt = (bid >> 3) & 31;   // HO/ROWS
  const int b   = bid >> 8;
  const int ox0 = xt * 16;
  const int oy0 = oyt * ROWS;

  __shared__ bf16_t smem[6 * 36 * 128];  // 55296 B

  const int tid = threadIdx.x;
  const int gx0 = 2 * ox0 - 1;
  const int gy0 = 2 * oy0 - 1;
  for (int i = tid; i < 6 * 36 * 128; i += 256) {
    const int col = i % 36;
    const int ch  = (i / 36) & 127;
    const int row = i / (36 * 128);
    const int gx = gx0 + col;
    const int gy = gy0 + row;
    float v = 0.0f;
    if (gx >= 0 && gx < W && gy >= 0 && gy < H)
      v = (float)xin[(((size_t)b * C + ch) * H + gy) * W + gx];
    smem[(row * 36 + col) * 128 + ch] = (bf16_t)v;
  }
  __syncthreads();

  const int wave = tid >> 5;
  const int lane = tid & 31;
  const int half = lane >> 4;
  const int mn   = lane & 15;
  const int m0   = mg * 128 + wave * 16;

  v8f acc[ROWS];
#pragma unroll
  for (int r = 0; r < ROWS; ++r) acc[r] = (v8f){};

  for (int t = 0; t < 16; ++t) {
    const int dy = t >> 2, dx = t & 3;
    const bf16_t* wr = wd + ((size_t)t * CO + (m0 + mn)) * 128;
#pragma unroll
    for (int ks = 0; ks < 4; ++ks) {
      const int ka = ks * 32 + half * 16;
      const int kb = ks * 32 + half * 16;
      const v16bf a = ld_a(wr + ka);
#pragma unroll
      for (int r = 0; r < ROWS; ++r) {
        // output row oy0+r reads input row 2*(oy0+r)-1+dy -> LDS row (2r+dy)
        const bf16_t* bp = &smem[((2 * r + dy) * 36 + (2 * mn + dx)) * 128];
        acc[r] = wmma_bf16(a, ld_b(bp + kb), acc[r]);
      }
    }
  }

#pragma unroll
  for (int r = 0; r < ROWS; ++r) {
    const int oy = oy0 + r;
#pragma unroll
    for (int e = 0; e < 8; ++e) {
      const int co = m0 + e + half * 8;
      const int px = ox0 + mn;
      out[(((size_t)b * CO + co) * HO + oy) * WO + px] = acc[r][e] + bias[co];
    }
  }
}

extern "C" void kernel_launch(void* const* d_in, const int* in_sizes, int n_in,
                              void* d_out, int out_size, void* d_ws, size_t ws_size,
                              hipStream_t stream) {
  (void)in_sizes; (void)n_in; (void)out_size; (void)ws_size;
  // setup_inputs order:
  // 0 features, 1 style_mean1, 2 style_std1, 3 style_mean2, 4 style_std2,
  // 5 lastBlockBool, 6 w1, 7 w2, 8 down_w, 9 down_b
  const float* feat = (const float*)d_in[0];
  const float* s1   = (const float*)d_in[1];   // reference: s1 = style_mean1
  const float* s2   = (const float*)d_in[2];   // reference: s2 = style_std1
  const float* w1   = (const float*)d_in[6];
  const float* w2   = (const float*)d_in[7];
  const float* dw   = (const float*)d_in[8];
  const float* db   = (const float*)d_in[9];
  float* out = (float*)d_out;

  // Workspace carve-out (~138 MB total).
  char* ws = (char*)d_ws;
  const size_t WMOD = (size_t)16 * 9 * 128 * 128;     // per-layer modulated weights
  const size_t XSZ  = (size_t)16 * 128 * 128 * 128;   // activation tensor
  bf16_t* wmod1 = (bf16_t*)ws;  ws += WMOD * sizeof(bf16_t);
  bf16_t* wmod2 = (bf16_t*)ws;  ws += WMOD * sizeof(bf16_t);
  bf16_t* dwbf  = (bf16_t*)ws;  ws += (size_t)16 * 256 * 128 * sizeof(bf16_t);
  bf16_t* x1    = (bf16_t*)ws;  ws += XSZ * sizeof(bf16_t);
  bf16_t* x2    = (bf16_t*)ws;  ws += XSZ * sizeof(bf16_t);

  modulate_kernel<<<16 * 128, 256, 0, stream>>>(w1, s1, wmod1);
  modulate_kernel<<<16 * 128, 256, 0, stream>>>(w2, s2, wmod2);
  cvt_downw_kernel<<<2048, 256, 0, stream>>>(dw, dwbf);

  // conv1: features(f32) -> x1 (bf16), lrelu. Grid: 16 * 32 * 4 = 2048.
  conv3x3_mod_kernel<true,  false><<<2048, 256, 0, stream>>>(feat, nullptr, wmod1, x1);
  // conv2: x1 -> x2, +residual (from LDS), lrelu.
  conv3x3_mod_kernel<false, true ><<<2048, 256, 0, stream>>>(nullptr, x1, wmod2, x2);
  // down: x2 -> out (f32) + bias. Grid: 16 * 32 * 4 * 2 = 4096.
  conv4x4s2_kernel<<<4096, 256, 0, stream>>>(x2, dwbf, db, out);
}
